// MultiheadedSelfAttention_85684597555489
// MI455X (gfx1250) — matrix operand order repaired and checked by reading, not compiled, not run on hardware
//
#include <hip/hip_runtime.h>
#include <hip/hip_bf16.h>

#define BB 2
#define SS 2048
#define EE 1024
#define HH 16
#define HD 64
#define MTOT (BB * SS) /* 4096 */
#define NEGV -1e9f

typedef __bf16 bf16_t;
typedef __attribute__((ext_vector_type(16))) bf16_t v16bf;
typedef __attribute__((ext_vector_type(8)))  bf16_t v8bf;
typedef __attribute__((ext_vector_type(8)))  float  v8f;
typedef __attribute__((ext_vector_type(4)))  float  v4f;

union ABu { v16bf v; v8bf h[2]; };

static __device__ __forceinline__ v8f wmma_bf16(v16bf a, v16bf b, v8f c) {
  // D = A(16x32 bf16) * B(32x16 bf16) + C(16x16 f32)
  return __builtin_amdgcn_wmma_f32_16x16x32_bf16(false, a, false, b, (short)0, c,
                                                 false, false);
}

// A-frag (16x32 bf16): lane<16 -> row M=lane, K={0..7,16..23};
// lane>=16 -> row M=lane-16, K={8..15,24..31}.
static __device__ __forceinline__ v16bf load_a16(const bf16_t* base, int stride, int lane) {
  int m = lane & 15, k0 = (lane >> 4) * 8;
  ABu u;
  u.h[0] = *(const v8bf*)(base + (size_t)m * stride + k0);
  u.h[1] = *(const v8bf*)(base + (size_t)m * stride + k0 + 16);
  return u.v;
}

// B-frag (32x16 bf16): lane<16 -> col N=lane, K={0..7,16..23};
// lane>=16 -> col N=lane-16, K={8..15,24..31}.
static __device__ __forceinline__ v16bf load_b16(const bf16_t* base, int stride, int lane) {
  int n = lane & 15, k0 = (lane >> 4) * 8;
  ABu u;
#pragma unroll
  for (int i = 0; i < 8; ++i) {
    u.h[0][i] = base[(size_t)(k0 + i) * stride + n];
    u.h[1][i] = base[(size_t)(k0 + 16 + i) * stride + n];
  }
  return u.v;
}

// ---------------------------------------------------------------------------
// Projection GEMM: out(B,H,S,HD) bf16 = X(f32, M x E) @ W^T(f32, E x E) + b
// block = 256 thr (8 waves); block tile 128 rows x 64 cols; wave = 16 rows.
// ---------------------------------------------------------------------------
__global__ __launch_bounds__(256) void proj_qkv(const float* __restrict__ X,
                                                const float* __restrict__ W,
                                                const float* __restrict__ bias,
                                                bf16_t* __restrict__ out) {
  __shared__ __align__(16) bf16_t Wt[32][80];  // Wt[k][n] = W[nb+n][kb+k]
  int lane = threadIdx.x & 31, wave = threadIdx.x >> 5;
  int nb = blockIdx.x * 64;
  int rb = blockIdx.y * 128 + wave * 16;
  v8f acc[4] = {};
  for (int kb = 0; kb < EE; kb += 32) {
    {  // stage transposed W tile, f32 -> bf16
      int n = threadIdx.x >> 2;        // 0..63
      int k0 = (threadIdx.x & 3) * 8;  // 0,8,16,24
      const float* wp = W + (size_t)(nb + n) * EE + kb + k0;
      v4f w0 = *(const v4f*)wp, w1 = *(const v4f*)(wp + 4);
#pragma unroll
      for (int i = 0; i < 4; ++i) {
        Wt[k0 + i][n]     = (bf16_t)w0[i];
        Wt[k0 + 4 + i][n] = (bf16_t)w1[i];
      }
    }
    __syncthreads();
    // A-frag straight from global (f32 -> bf16)
    int m = lane & 15, kk = (lane >> 4) * 8;
    const float* xp = X + (size_t)(rb + m) * EE + kb + kk;
    v4f a0 = *(const v4f*)xp,        a1 = *(const v4f*)(xp + 4);
    v4f a2 = *(const v4f*)(xp + 16), a3 = *(const v4f*)(xp + 20);
    ABu ua;
#pragma unroll
    for (int i = 0; i < 4; ++i) {
      ua.h[0][i]     = (bf16_t)a0[i];
      ua.h[0][4 + i] = (bf16_t)a1[i];
      ua.h[1][i]     = (bf16_t)a2[i];
      ua.h[1][4 + i] = (bf16_t)a3[i];
    }
#pragma unroll
    for (int t = 0; t < 4; ++t)
      acc[t] = wmma_bf16(ua.v, load_b16(&Wt[0][t * 16], 80, lane), acc[t]);
    __syncthreads();
  }
  // epilogue: + bias, scatter to (B,H,S,HD) bf16
  int half = lane >> 4, nl = lane & 15;
#pragma unroll
  for (int t = 0; t < 4; ++t) {
    int n = nb + t * 16 + nl;
    float bv = bias[n];
    int h = n >> 6, d = n & 63;
#pragma unroll
    for (int r = 0; r < 8; ++r) {
      int row = rb + r + 8 * half;
      int b = row >> 11, s = row & (SS - 1);
      out[(((size_t)b * HH + h) * SS + s) * HD + d] = (bf16_t)(acc[t][r] + bv);
    }
  }
}

// ---------------------------------------------------------------------------
// Output projection: out(f32, M x E) = ctx(bf16) @ Wo^T + bo
// ---------------------------------------------------------------------------
__global__ __launch_bounds__(256) void out_proj(const bf16_t* __restrict__ Xc,
                                                const float* __restrict__ W,
                                                const float* __restrict__ bias,
                                                float* __restrict__ out) {
  __shared__ __align__(16) bf16_t Wt[32][80];
  int lane = threadIdx.x & 31, wave = threadIdx.x >> 5;
  int nb = blockIdx.x * 64;
  int rb = blockIdx.y * 128 + wave * 16;
  v8f acc[4] = {};
  for (int kb = 0; kb < EE; kb += 32) {
    int n = threadIdx.x >> 2, k0 = (threadIdx.x & 3) * 8;
    const float* wp = W + (size_t)(nb + n) * EE + kb + k0;
    v4f w0 = *(const v4f*)wp, w1 = *(const v4f*)(wp + 4);
#pragma unroll
    for (int i = 0; i < 4; ++i) {
      Wt[k0 + i][n]     = (bf16_t)w0[i];
      Wt[k0 + 4 + i][n] = (bf16_t)w1[i];
    }
    __syncthreads();
    v16bf a = load_a16(Xc + (size_t)rb * EE + kb, EE, lane);
#pragma unroll
    for (int t = 0; t < 4; ++t)
      acc[t] = wmma_bf16(a, load_b16(&Wt[0][t * 16], 80, lane), acc[t]);
    __syncthreads();
  }
  int half = lane >> 4, nl = lane & 15;
#pragma unroll
  for (int t = 0; t < 4; ++t) {
    int n = nb + t * 16 + nl;
    float bv = bias[n];
#pragma unroll
    for (int r = 0; r < 8; ++r) {
      int row = rb + r + 8 * half;
      out[(size_t)row * EE + n] = acc[t][r] + bv;
    }
  }
}

// ---------------------------------------------------------------------------
// Fused causal attention per (b*h, 64-row q-block); 4 waves x 16 q-rows.
// pass1: online (rowmax, rowsum) over causal keys via WMMA QK^T.
// pass2: recompute scores, normalize, single write of attn, P@V via WMMA.
// ---------------------------------------------------------------------------
__global__ __launch_bounds__(128) void attention(const bf16_t* __restrict__ qws,
                                                 const bf16_t* __restrict__ kws,
                                                 const bf16_t* __restrict__ vws,
                                                 float* __restrict__ attn,
                                                 bf16_t* __restrict__ ctx) {
  __shared__ __align__(16) bf16_t Kt[64][80];      // Kt[d][key] (transposed K tile)
  __shared__ __align__(16) bf16_t Pb[4][16][80];   // per-wave P row-major buffer
  int lane = threadIdx.x & 31, wave = threadIdx.x >> 5;
  int qb = blockIdx.x;  // 64-row q block
  int bh = blockIdx.y;  // b*H + h
  const bf16_t* qp = qws + (size_t)bh * SS * HD;
  const bf16_t* kp = kws + (size_t)bh * SS * HD;
  const bf16_t* vp = vws + (size_t)bh * SS * HD;
  float* ap = attn + (size_t)bh * SS * SS;
  int qr0 = qb * 64 + wave * 16;
  int half = lane >> 4, nl = lane & 15;

  v16bf aq0 = load_a16(qp + (size_t)qr0 * HD + 0,  HD, lane);
  v16bf aq1 = load_a16(qp + (size_t)qr0 * HD + 32, HD, lane);

  float rmax[8], rsum[8];
#pragma unroll
  for (int r = 0; r < 8; ++r) { rmax[r] = -3.0e38f; rsum[r] = 0.f; }

  // ---------------- pass 1: row max / sum(exp) ----------------
  for (int kb = 0; kb <= qb; ++kb) {
    for (int i = threadIdx.x; i < 64 * 64; i += 128) {
      int key = i >> 6, d = i & 63;
      Kt[d][key] = kp[(size_t)(kb * 64 + key) * HD + d];
    }
    __syncthreads();
    v8f sc[4];
#pragma unroll
    for (int t = 0; t < 4; ++t) {
      v8f c = {};
      c = wmma_bf16(aq0, load_b16(&Kt[0][t * 16],  80, lane), c);
      c = wmma_bf16(aq1, load_b16(&Kt[32][t * 16], 80, lane), c);
#pragma unroll
      for (int r = 0; r < 8; ++r) c[r] *= 0.125f;  // 1/sqrt(HD)
      if (kb == qb) {
        int key = kb * 64 + t * 16 + nl;
#pragma unroll
        for (int r = 0; r < 8; ++r) {
          int q = qr0 + r + 8 * half;
          if (key > q) c[r] = NEGV;
        }
      }
      sc[t] = c;
    }
#pragma unroll
    for (int r = 0; r < 8; ++r) {
      float v = fmaxf(fmaxf(sc[0][r], sc[1][r]), fmaxf(sc[2][r], sc[3][r]));
#pragma unroll
      for (int off = 8; off >= 1; off >>= 1) v = fmaxf(v, __shfl_xor(v, off, 16));
      float nm = fmaxf(rmax[r], v);
      float p = __expf(sc[0][r] - nm) + __expf(sc[1][r] - nm) +
                __expf(sc[2][r] - nm) + __expf(sc[3][r] - nm);
#pragma unroll
      for (int off = 8; off >= 1; off >>= 1) p += __shfl_xor(p, off, 16);
      rsum[r] = rsum[r] * __expf(rmax[r] - nm) + p;
      rmax[r] = nm;
    }
    __syncthreads();
  }
  float rinv[8];
#pragma unroll
  for (int r = 0; r < 8; ++r) rinv[r] = 1.0f / rsum[r];

  // ---------------- pass 2: normalize + write attn + P@V ----------------
  v8f acc[4] = {};
  for (int kb = 0; kb <= qb; ++kb) {
    for (int i = threadIdx.x; i < 64 * 64; i += 128) {
      int key = i >> 6, d = i & 63;
      Kt[d][key] = kp[(size_t)(kb * 64 + key) * HD + d];
    }
    __syncthreads();
#pragma unroll
    for (int t = 0; t < 4; ++t) {
      v8f c = {};
      c = wmma_bf16(aq0, load_b16(&Kt[0][t * 16],  80, lane), c);
      c = wmma_bf16(aq1, load_b16(&Kt[32][t * 16], 80, lane), c);
      int key = kb * 64 + t * 16 + nl;
#pragma unroll
      for (int r = 0; r < 8; ++r) {
        int q = qr0 + r + 8 * half;
        float s = c[r] * 0.125f;
        if (kb == qb && key > q) s = NEGV;         // exp underflows to exactly 0
        float p = __expf(s - rmax[r]) * rinv[r];
        ap[(size_t)q * SS + key] = p;              // the one-and-only attn write
        Pb[wave][r + 8 * half][t * 16 + nl] = (bf16_t)p;
      }
    }
    // same-wave LDS store->load; DS ops are in-order per wave
    v16bf pa0 = load_a16(&Pb[wave][0][0],  80, lane);
    v16bf pa1 = load_a16(&Pb[wave][0][32], 80, lane);
#pragma unroll
    for (int t = 0; t < 4; ++t) {
      acc[t] = wmma_bf16(pa0, load_b16(vp + (size_t)(kb * 64) * HD + t * 16,      HD, lane), acc[t]);
      acc[t] = wmma_bf16(pa1, load_b16(vp + (size_t)(kb * 64 + 32) * HD + t * 16, HD, lane), acc[t]);
    }
    __syncthreads();
  }
  // write context (B,S,E) bf16, heads interleaved
  int b = bh >> 4, h = bh & 15;
#pragma unroll
  for (int t = 0; t < 4; ++t)
#pragma unroll
    for (int r = 0; r < 8; ++r) {
      int q = qr0 + r + 8 * half;
      int d = t * 16 + nl;
      ctx[((size_t)b * SS + q) * EE + h * HD + d] = (bf16_t)acc[t][r];
    }
  // zero-fill the fully-masked upper-triangle blocks (d_out is poisoned)
  int zc = SS - (qb + 1) * 64;
  if (zc > 0) {
    int zc4 = zc >> 2;
    float* base = ap + (size_t)(qb * 64) * SS + (qb + 1) * 64;
    for (int i = threadIdx.x; i < 64 * zc4; i += 128) {
      int row = i / zc4, c4 = i - row * zc4;
      v4f z = {};
      *(v4f*)(base + (size_t)row * SS + (size_t)c4 * 4) = z;
    }
  }
}

// ---------------------------------------------------------------------------
extern "C" void kernel_launch(void* const* d_in, const int* in_sizes, int n_in,
                              void* d_out, int out_size, void* d_ws, size_t ws_size,
                              hipStream_t stream) {
  const float* query = (const float*)d_in[0];
  const float* key   = (const float*)d_in[1];
  const float* value = (const float*)d_in[2];
  /* d_in[3] = causal mask (int), implied by the kernel's triangular loop */
  const float* Wq = (const float*)d_in[4];
  const float* bq = (const float*)d_in[5];
  const float* Wk = (const float*)d_in[6];
  const float* bk = (const float*)d_in[7];
  const float* Wv = (const float*)d_in[8];
  const float* bv = (const float*)d_in[9];
  const float* Wo = (const float*)d_in[10];
  const float* bo = (const float*)d_in[11];

  float* out  = (float*)d_out;                  // (B,S,E), first output
  float* attn = out + (size_t)BB * SS * EE;     // (B,H,S,S), second output

  const size_t perT = (size_t)BB * HH * SS * HD;  // 4M bf16 = 8 MB each
  bf16_t* qws = (bf16_t*)d_ws;
  bf16_t* kws = qws + perT;
  bf16_t* vws = kws + perT;
  bf16_t* ctx = vws + perT;                     // total ws use: 32 MB

  dim3 gproj(EE / 64, MTOT / 128);
  proj_qkv<<<gproj, 256, 0, stream>>>(query, Wq, bq, qws);
  proj_qkv<<<gproj, 256, 0, stream>>>(key,   Wk, bk, kws);
  proj_qkv<<<gproj, 256, 0, stream>>>(value, Wv, bv, vws);

  dim3 gattn(SS / 64, BB * HH);
  attention<<<gattn, 128, 0, stream>>>(qws, kws, vws, attn, ctx);

  out_proj<<<gproj, 256, 0, stream>>>(ctx, Wo, bo, out);
}
